// ESRNN_26156350832961
// MI455X (gfx1250) — matrix operand order repaired and checked by reading, not compiled, not run on hardware
//
#include <hip/hip_runtime.h>
#include <hip/hip_bf16.h>
#include <math.h>

// ES-RNN forward for MI455X (gfx1250, wave32).
// Strategy: recurrence-latency-bound workload (~1864 dependent GEMM steps,
// ~0.5 TFLOP total, working set << 192MB L2). Gate GEMMs run on
// v_wmma_f32_16x16x32_bf16 inside a single persistent workgroup per LSTM
// cell; bf16 storage halves L2 traffic; f32 accumulation preserves the
// recurrent state. Scalar ES scan / window build / collapsed head are VALU.

typedef unsigned short u16;
typedef unsigned int   u32;
typedef __attribute__((ext_vector_type(16))) __bf16 v16bf;
typedef __attribute__((ext_vector_type(8)))  float  v8f;

#define T_LEN 1024
#define N_SER 128
#define NSEAS 7
#define IWIN  32
#define HID   256
#define GDIM  1024          // 4*H gate width
#define NWIN  993           // T - IW + 1
#define F0PAD 64            // cell0 input features 33 -> padded to 64 (2 K-steps)

__device__ __forceinline__ u16 f2bf(float f) {
  u32 u = __float_as_uint(f);
  u32 r = u + 0x7FFFu + ((u >> 16) & 1u);   // round-to-nearest-even
  return (u16)(r >> 16);
}
__device__ __forceinline__ float bf2f(u16 b) {
  return __uint_as_float(((u32)b) << 16);
}
__device__ __forceinline__ float sigmoidf(float x) { return 1.0f / (1.0f + expf(-x)); }

// ---------------- weight prep ----------------
__global__ void pad_bf16_kernel(u16* dst, const float* src, int rows, int sc, int dc) {
  int idx = blockIdx.x * blockDim.x + threadIdx.x;
  if (idx >= rows * dc) return;
  int r = idx / dc, c = idx % dc;
  dst[idx] = (c < sc) ? f2bf(src[r * sc + c]) : (u16)0;
}

__global__ void bias_kernel(float* dst, const float* a, const float* b, int n) {
  int i = blockIdx.x * blockDim.x + threadIdx.x;
  if (i < n) dst[i] = a[i] + b[i];
}

// ---------------- exponential-smoothing scan (1 thread per series) ----------------
__global__ void es_scan_kernel(const float* __restrict__ x,
                               const float* __restrict__ ilev,
                               const float* __restrict__ isms,
                               const float* __restrict__ iseas,
                               float* __restrict__ levs,   // [N][T]
                               float* __restrict__ seas) { // [N][T] (seas_stacked[:, 0:T])
  int n = threadIdx.x;
  if (n >= N_SER) return;
  float lsm = sigmoidf(ilev[n]);
  float ssm = sigmoidf(isms[n]);
  float ring[NSEAS];
  #pragma unroll
  for (int i = 0; i < NSEAS; ++i) ring[i] = expf(iseas[n * NSEAS + i]);
  float lev = x[n] / ring[0];               // x[0][n]
  levs[n * T_LEN] = lev;
  #pragma unroll
  for (int i = 0; i < NSEAS; ++i) seas[n * T_LEN + i] = ring[i];
  seas[n * T_LEN + 7] = ring[0];            // seas_init_full = [ring0, ring0[0]]
  for (int s = 1; s < T_LEN; ++s) {
    int p = s % NSEAS;
    float xs = x[s * N_SER + n];
    float ss = ring[p];
    float nlv = lsm * (xs / ss) + (1.0f - lsm) * lev;
    float ns  = ssm * (xs / nlv) + (1.0f - ssm) * ss;
    ring[p] = ns;
    levs[n * T_LEN + s] = nlv;
    if (s + 7 < T_LEN) seas[n * T_LEN + s + 7] = ns;  // seas_y[s-1] -> index s+7
    lev = nlv;
  }
}

// ---------------- window build: win[w][n][f] bf16, F padded to 64 ----------------
__global__ void build_win_kernel(const float* __restrict__ x,
                                 const float* __restrict__ levs,
                                 const float* __restrict__ seas,
                                 u16* __restrict__ win) {
  int idx = blockIdx.x * blockDim.x + threadIdx.x;
  if (idx >= NWIN * N_SER * F0PAD) return;
  int f  = idx & (F0PAD - 1);
  int rn = idx >> 6;
  int n  = rn & (N_SER - 1);
  int w  = rn >> 7;
  float v;
  if (f < IWIN) {
    int t = w + f;
    float xv = x[t * N_SER + n];
    float s  = seas[n * T_LEN + t];
    float l  = levs[n * T_LEN + w + IWIN - 1];
    v = xv / s / l;
    if (isnan(v)) v = 0.0f;
    else if (isinf(v)) v = (v > 0.0f) ? 3.4028235e38f : -3.4028235e38f;
  } else {
    v = (f == IWIN) ? 1.0f : 0.0f;          // appended ones column + zero pad
  }
  win[idx] = f2bf(v);
}

// ---------------- persistent dilated-LSTM cell (single workgroup, 8 waves) ----------------
// gates = X @ Wih^T + Hprev @ Whh^T + bias  via v_wmma_f32_16x16x32_bf16,
// then elementwise LSTM update; workgroup barrier + device fence between
// chunks resolves the recurrence without kernel relaunch.
__launch_bounds__(256)
__global__ void lstm_cell_kernel(const u16* Xin, int Fx,
                                 const u16* __restrict__ Wih, const u16* __restrict__ Whh,
                                 const float* __restrict__ bias,
                                 u16* Hout, const u16* resid,
                                 int rate,
                                 u16* hstate, float* cstate, float* gates) {
  const int tid  = threadIdx.x;
  const int lane = tid & 31;
  const int wave = tid >> 5;
  const int hl   = lane >> 4;     // half: 0 = lanes 0-15, 1 = lanes 16-31
  const int ln   = lane & 15;
  const int rB      = rate * N_SER;
  const int nTiles  = (rB >> 4) * (GDIM / 16);
  const int nChunks = (NWIN + rate - 1) / rate;
  const size_t totRows = (size_t)NWIN * N_SER;

  const __bf16* Wi = (const __bf16*)Wih;
  const __bf16* Wh = (const __bf16*)Whh;
  const __bf16* Hb = (const __bf16*)hstate;

  for (int i = tid; i < rB * HID; i += 256) { hstate[i] = 0; cstate[i] = 0.0f; }
  __threadfence();
  __syncthreads();

  for (int k = 0; k < nChunks; ++k) {
    const size_t chunkRow0 = (size_t)k * rB;
    const __bf16* Xb = (const __bf16*)Xin + chunkRow0 * (size_t)Fx;

    // ---- phase 1: WMMA gate GEMM, tiles striped over the 8 waves ----
    for (int tile = wave; tile < nTiles; tile += 8) {
      int rt = tile >> 6;
      int ct = tile & 63;
      int mr = rt << 4;
      int nc = ct << 4;
      v8f acc;
      #pragma unroll
      for (int i = 0; i < 8; ++i) acc[i] = 0.0f;

      int arow = mr + ln;
      bool avalid = (chunkRow0 + (size_t)arow) < totRows;   // zero-pad tail chunk

      for (int kk = 0; kk < Fx; kk += 32) {                 // X @ Wih^T
        v16bf a, b;
        #pragma unroll
        for (int p = 0; p < 8; ++p) {
          int ka = ((p < 4) ? (2 * p) : (8 + 2 * p)) + 8 * hl + kk;   // A 16x32 layout
          a[2*p]   = avalid ? Xb[(size_t)arow * Fx + ka]     : (__bf16)0.0f;
          a[2*p+1] = avalid ? Xb[(size_t)arow * Fx + ka + 1] : (__bf16)0.0f;
          int kb = ((p < 4) ? (2 * p) : (8 + 2 * p)) + 8 * hl + kk;   // B 32x16 layout
          b[2*p]   = Wi[(size_t)(nc + ln) * Fx + kb];
          b[2*p+1] = Wi[(size_t)(nc + ln) * Fx + kb + 1];
        }
        acc = __builtin_amdgcn_wmma_f32_16x16x32_bf16(false, a, false, b,
                                                      (short)0, acc, false, false);
      }
      #pragma unroll
      for (int kk = 0; kk < HID; kk += 32) {                // Hprev @ Whh^T
        v16bf a, b;
        #pragma unroll
        for (int p = 0; p < 8; ++p) {
          int ka = ((p < 4) ? (2 * p) : (8 + 2 * p)) + 8 * hl + kk;
          a[2*p]   = Hb[(size_t)arow * HID + ka];
          a[2*p+1] = Hb[(size_t)arow * HID + ka + 1];
          int kb = ((p < 4) ? (2 * p) : (8 + 2 * p)) + 8 * hl + kk;
          b[2*p]   = Wh[(size_t)(nc + ln) * HID + kb];
          b[2*p+1] = Wh[(size_t)(nc + ln) * HID + kb + 1];
        }
        acc = __builtin_amdgcn_wmma_f32_16x16x32_bf16(false, a, false, b,
                                                      (short)0, acc, false, false);
      }
      float bv = bias[nc + ln];
      #pragma unroll
      for (int i = 0; i < 8; ++i) {                         // D layout: M=i+8*hl, N=ln
        int row = mr + i + 8 * hl;
        gates[(size_t)row * GDIM + nc + ln] = acc[i] + bv;
      }
    }
    __threadfence();
    __syncthreads();

    // ---- phase 2: elementwise LSTM update + output scatter (+ residual) ----
    for (int idx = tid; idx < rB * HID; idx += 256) {
      int row = idx >> 8;
      int h   = idx & 255;
      const float* g = gates + (size_t)row * GDIM;
      float ig = sigmoidf(g[h]);
      float fg = sigmoidf(g[HID + h]);
      float gg = tanhf(g[2 * HID + h]);
      float og = sigmoidf(g[3 * HID + h]);
      float c  = fg * cstate[idx] + ig * gg;
      float hv = og * tanhf(c);
      cstate[idx] = c;
      hstate[idx] = f2bf(hv);
      size_t grow = chunkRow0 + (size_t)row;
      if (grow < totRows) {
        float o = hv;
        if (resid) o += bf2f(resid[grow * HID + h]);
        Hout[grow * HID + h] = f2bf(o);
      }
      if (k + 1 < nChunks) {                                // warm L2/L0 for next chunk
        size_t nrow = chunkRow0 + (size_t)rB + (size_t)row;
        if (nrow < totRows) __builtin_prefetch(Xin + nrow * Fx, 0, 0);
      }
    }
    __threadfence();
    __syncthreads();
  }
}

// ---------------- head: tanh(H @ nl_w^T + nl_b) -> relu(row1 of sc) -> fc ----------------
__global__ void head_kernel(const u16* __restrict__ H3,
                            const float* __restrict__ nl_w, const float* __restrict__ nl_b,
                            const float* __restrict__ sc_w, const float* __restrict__ sc_b,
                            const float* __restrict__ fc_w, const float* __restrict__ fc_b,
                            float* __restrict__ out) {
  int w = blockIdx.x;
  int n = threadIdx.x;                          // 128 threads = one per series
  const u16* hrow = H3 + ((size_t)w * N_SER + n) * HID;
  float msum = 0.0f;
  for (int j = 0; j < HID; ++j) {
    float acc = nl_b[j];
    const float* wr = nl_w + (size_t)j * HID;
    for (int kk = 0; kk < HID; ++kk) acc += bf2f(hrow[kk]) * wr[kk];
    msum += tanhf(acc) * sc_w[HID + j];         // only scoring row 1 survives
  }
  float m = msum + sc_b[1];
  m = (m > 0.0f) ? m : 0.0f;                    // relu
  __shared__ float red[N_SER];
  red[n] = m * fc_w[n];
  __syncthreads();
  for (int s = N_SER / 2; s > 0; s >>= 1) {
    if (n < s) red[n] += red[n + s];
    __syncthreads();
  }
  if (n == 0) out[(T_LEN - NWIN) + w] = red[0] + fc_b[0];
}

__global__ void fill_pad_kernel(float* out) {
  int i = threadIdx.x;
  float v = out[T_LEN - NWIN];                  // pad_val = out[0] of the unpadded seq
  if (i < T_LEN - NWIN) out[i] = v;
}

// ---------------- host orchestration ----------------
extern "C" void kernel_launch(void* const* d_in, const int* in_sizes, int n_in,
                              void* d_out, int out_size, void* d_ws, size_t ws_size,
                              hipStream_t stream) {
  (void)in_sizes; (void)n_in; (void)out_size; (void)ws_size;
  const float* x       = (const float*)d_in[0];
  const float* ilev    = (const float*)d_in[1];
  const float* isms    = (const float*)d_in[2];
  const float* iseason = (const float*)d_in[3];
  const float* fc_w    = (const float*)d_in[4];
  const float* fc_b    = (const float*)d_in[5];
  const float* nl_w    = (const float*)d_in[6];
  const float* nl_b    = (const float*)d_in[7];
  const float* sc_w    = (const float*)d_in[8];
  const float* sc_b    = (const float*)d_in[9];
  // cells flattened: d_in[10 + 4*c + {0:Wih, 1:Whh, 2:bih, 3:bhh}]

  char* ws = (char*)d_ws;
  size_t off = 0;
  auto alloc = [&](size_t bytes) -> char* {
    off = (off + 255) & ~(size_t)255;
    char* p = ws + off;
    off += bytes;
    return p;
  };

  float* levs = (float*)alloc(sizeof(float) * (size_t)N_SER * T_LEN);
  float* seas = (float*)alloc(sizeof(float) * (size_t)N_SER * T_LEN);
  u16*   win  = (u16*)  alloc(sizeof(u16)   * (size_t)NWIN * N_SER * F0PAD);

  const int Fx[4]   = {F0PAD, HID, HID, HID};
  const int Fsrc[4] = {IWIN + 1, HID, HID, HID};
  u16* wih[4]; u16* whh[4]; float* bias[4];
  for (int c = 0; c < 4; ++c) {
    wih[c]  = (u16*)  alloc(sizeof(u16)   * (size_t)GDIM * Fx[c]);
    whh[c]  = (u16*)  alloc(sizeof(u16)   * (size_t)GDIM * HID);
    bias[c] = (float*)alloc(sizeof(float) * GDIM);
  }
  u16*   bufA   = (u16*)  alloc(sizeof(u16)   * (size_t)NWIN * N_SER * HID);
  u16*   bufB   = (u16*)  alloc(sizeof(u16)   * (size_t)NWIN * N_SER * HID);
  float* gates  = (float*)alloc(sizeof(float) * (size_t)1024 * GDIM);
  u16*   hstate = (u16*)  alloc(sizeof(u16)   * (size_t)1024 * HID);
  float* cstate = (float*)alloc(sizeof(float) * (size_t)1024 * HID);

  // weight prep (bf16 conversion + zero-pad cell0 features 33->64, fuse biases)
  for (int c = 0; c < 4; ++c) {
    int tot = GDIM * Fx[c];
    pad_bf16_kernel<<<(tot + 255) / 256, 256, 0, stream>>>(
        wih[c], (const float*)d_in[10 + 4 * c + 0], GDIM, Fsrc[c], Fx[c]);
    tot = GDIM * HID;
    pad_bf16_kernel<<<(tot + 255) / 256, 256, 0, stream>>>(
        whh[c], (const float*)d_in[10 + 4 * c + 1], GDIM, HID, HID);
    bias_kernel<<<(GDIM + 255) / 256, 256, 0, stream>>>(
        bias[c], (const float*)d_in[10 + 4 * c + 2], (const float*)d_in[10 + 4 * c + 3], GDIM);
  }

  es_scan_kernel<<<1, 128, 0, stream>>>(x, ilev, isms, iseason, levs, seas);

  int wtot = NWIN * N_SER * F0PAD;
  build_win_kernel<<<(wtot + 255) / 256, 256, 0, stream>>>(x, levs, seas, win);

  // dilated LSTM stack (dils [1,2],[4,8]); residual from cell1 output after cell3
  lstm_cell_kernel<<<1, 256, 0, stream>>>(win,  F0PAD, wih[0], whh[0], bias[0], bufA, nullptr, 1, hstate, cstate, gates);
  lstm_cell_kernel<<<1, 256, 0, stream>>>(bufA, HID,   wih[1], whh[1], bias[1], bufB, nullptr, 2, hstate, cstate, gates);
  lstm_cell_kernel<<<1, 256, 0, stream>>>(bufB, HID,   wih[2], whh[2], bias[2], bufA, nullptr, 4, hstate, cstate, gates);
  lstm_cell_kernel<<<1, 256, 0, stream>>>(bufA, HID,   wih[3], whh[3], bias[3], bufA, bufB,   8, hstate, cstate, gates);

  head_kernel<<<NWIN, 128, 0, stream>>>(bufA, nl_w, nl_b, sc_w, sc_b, fc_w, fc_b, (float*)d_out);
  fill_pad_kernel<<<1, 64, 0, stream>>>((float*)d_out);
}
